// NeuralODE_27831388078723
// MI455X (gfx1250) — matrix operand order repaired and checked by reading, not compiled, not run on hardware
//
#include <hip/hip_runtime.h>
#include <hip/hip_bf16.h>
#include <math.h>

// ---------- Tsit5 coefficients ----------
#define A21 0.161f
#define A31 (-0.008480655492356989f)
#define A32 0.335480655492357f
#define A41 2.8971530571054935f
#define A42 (-6.359448489975075f)
#define A43 4.3622954328695815f
#define A51 5.325864828439257f
#define A52 (-11.748883564062828f)
#define A53 7.4955393428898365f
#define A54 (-0.09249506636175525f)
#define A61 5.86145544294642f
#define A62 (-12.92096931784711f)
#define A63 8.159367898576159f
#define A64 (-0.071584973281401f)
#define A65 (-0.028269050394068383f)
#define B1c 0.09646076681806523f
#define B2c 0.01f
#define B3c 0.4798896504144996f
#define B4c 1.379008574103742f
#define B5c (-3.290069515436081f)
#define B6c 2.324710524099774f

#define DIM   64
#define WIDTH 256
#define BATCH 4096
#define NT    101

#if __has_builtin(__builtin_amdgcn_tanhf)
#define TANHF(x) __builtin_amdgcn_tanhf(x)
#else
#define TANHF(x) tanhf(x)
#endif

typedef __attribute__((ext_vector_type(16))) _Float16 v16h;
typedef __attribute__((ext_vector_type(8)))  float    v8f;
typedef __attribute__((ext_vector_type(4)))  unsigned int v4u;

union HV { v16h h; v4u u[2]; };

// wt layout (halfs): Wt0[256][64] | Wt1[256][256] | Wt2[256][256] | Wt3[64][256]
#define OFF_WT1 16384
#define OFF_WT2 81920
#define OFF_WT3 147456
#define WT_TOTAL 163840

// ---------------------------------------------------------------------------
// Prep: convert f32 weights to f16, transposed to [N][K] so that each lane's
// 32-K B-operand slice is contiguous (2x global_load_b128).
// ---------------------------------------------------------------------------
__global__ void prep_weights_kernel(const float* __restrict__ W0,
                                    const float* __restrict__ W1,
                                    const float* __restrict__ W2,
                                    const float* __restrict__ W3,
                                    _Float16* __restrict__ wt)
{
    int i = blockIdx.x * blockDim.x + threadIdx.x;
    if (i >= WT_TOTAL) return;
    if (i < OFF_WT1) {                     // Wt0[n][k] = W0[k][n], 256x64
        int n = i >> 6, k = i & 63;
        wt[i] = (_Float16)W0[k * WIDTH + n];
    } else if (i < OFF_WT2) {              // Wt1[n][k] = W1[k][n], 256x256
        int j = i - OFF_WT1; int n = j >> 8, k = j & 255;
        wt[i] = (_Float16)W1[k * WIDTH + n];
    } else if (i < OFF_WT3) {              // Wt2
        int j = i - OFF_WT2; int n = j >> 8, k = j & 255;
        wt[i] = (_Float16)W2[k * WIDTH + n];
    } else {                               // Wt3[n][k] = W3[k][n], 64x256
        int j = i - OFF_WT3; int n = j >> 8, k = j & 255;
        wt[i] = (_Float16)W3[k * DIM + n];
    }
}

// ---------------------------------------------------------------------------
// WMMA operand loaders.
// A (16x32 f16, M x K): lane l holds row M=l%16; each half-span of K is
// 8 contiguous halfs = 16B -> ds_load_b128 x2.
// B (32x16 f16, K x N): lane l holds col N=l%16; same K split as A (2x 16B of W^T row).
// ---------------------------------------------------------------------------
__device__ __forceinline__ v16h load_a_tile(const _Float16* __restrict__ act,
                                            int stride, int mBase, int kb,
                                            int lo, int hiA)
{
    HV a;
    const _Float16* p = act + (mBase + lo) * stride + kb + hiA;
    a.u[0] = *(const v4u*)p;
    a.u[1] = *(const v4u*)(p + 16);
    return a.h;
}

__device__ __forceinline__ v16h load_b_tile(const _Float16* __restrict__ Wt,
                                            int K, int nBase, int kb,
                                            int lo, int hiB)
{
    HV b;
    const _Float16* p = Wt + (nBase + lo) * K + kb + (hiB >> 1);
    b.u[0] = *(const v4u*)p;
    b.u[1] = *(const v4u*)(p + 16);
    return b.h;
}

// ---------------------------------------------------------------------------
// Hidden layer: actIn[64 x KIN] (LDS f16) -> actOut[64 x 256] (LDS f16),
// bias + tanh. Wave w computes columns [32w, 32w+32) for all 4 row tiles.
// Per k-step: burst-load all operands (2 global B + 4 LDS A tiles, one wait),
// then 8 independent back-to-back WMMAs.
// ---------------------------------------------------------------------------
__device__ __forceinline__ void layer_hidden(const _Float16* __restrict__ actIn,
                                             int sIn, int KIN,
                                             _Float16* __restrict__ actOut,
                                             const _Float16* __restrict__ Wt,
                                             float bv0, float bv1,
                                             int wave, int lane)
{
    const int lo = lane & 15;
    const int hiSel = (lane >> 4) & 1;
    const int hiA = hiSel * 8;
    const int hiB = hiSel * 16;
    const int rowOff = hiSel * 8;
    const int n0 = wave * 32;
    const int ktiles = KIN >> 5;

    v8f acc[4][2];
#pragma unroll
    for (int m = 0; m < 4; ++m) { acc[m][0] = (v8f){}; acc[m][1] = (v8f){}; }

    for (int kt = 0; kt < ktiles; ++kt) {
        const int kb = kt * 32;
        const v16h b0 = load_b_tile(Wt, KIN, n0,      kb, lo, hiB);
        const v16h b1 = load_b_tile(Wt, KIN, n0 + 16, kb, lo, hiB);
        v16h a[4];
#pragma unroll
        for (int m = 0; m < 4; ++m)
            a[m] = load_a_tile(actIn, sIn, m * 16, kb, lo, hiA);
#pragma unroll
        for (int m = 0; m < 4; ++m) {
            acc[m][0] = __builtin_amdgcn_wmma_f32_16x16x32_f16(
                false, a[m], false, b0, (short)0, acc[m][0], false, false);
            acc[m][1] = __builtin_amdgcn_wmma_f32_16x16x32_f16(
                false, a[m], false, b1, (short)0, acc[m][1], false, false);
        }
    }

#pragma unroll
    for (int m = 0; m < 4; ++m) {
#pragma unroll
        for (int r = 0; r < 8; ++r) {
            const int row = m * 16 + r + rowOff;
            actOut[row * 256 + n0 + lo]      = (_Float16)TANHF(acc[m][0][r] + bv0);
            actOut[row * 256 + n0 + 16 + lo] = (_Float16)TANHF(acc[m][1][r] + bv1);
        }
    }
}

// ---------------------------------------------------------------------------
// Output layer: actIn[64 x 256] -> 64x64 f32 written to an LDS k-buffer.
// Wave w owns tile pairs 2w, 2w+1; both share the same m row-tile, so one
// A load feeds two WMMAs per k-step. Write offsets are off64[].
// ---------------------------------------------------------------------------
__device__ __forceinline__ void layer_out(const _Float16* __restrict__ actIn,
                                          const _Float16* __restrict__ Wt3,
                                          float bvA, float bvB,
                                          int wave, int lane,
                                          float* __restrict__ kdst,
                                          const int off64[16])
{
    const int lo = lane & 15;
    const int hiSel = (lane >> 4) & 1;
    const int hiA = hiSel * 8;
    const int hiB = hiSel * 16;
    const int m  = (wave * 2) >> 2;        // shared row tile
    const int nA = (wave * 2) & 3;
    const int nB = nA + 1;

    v8f accA = {};
    v8f accB = {};
    for (int kt = 0; kt < 8; ++kt) {
        const int kb = kt * 32;
        const v16h bA = load_b_tile(Wt3, 256, nA * 16, kb, lo, hiB);
        const v16h bB = load_b_tile(Wt3, 256, nB * 16, kb, lo, hiB);
        const v16h a  = load_a_tile(actIn, 256, m * 16, kb, lo, hiA);
        accA = __builtin_amdgcn_wmma_f32_16x16x32_f16(
            false, a, false, bA, (short)0, accA, false, false);
        accB = __builtin_amdgcn_wmma_f32_16x16x32_f16(
            false, a, false, bB, (short)0, accB, false, false);
    }
#pragma unroll
    for (int r = 0; r < 8; ++r) {
        kdst[off64[r]]     = accA[r] + bvA;
        kdst[off64[8 + r]] = accB[r] + bvB;
    }
}

// ---------------------------------------------------------------------------
// Persistent ODE kernel: each block owns 64 batch rows for all 100 steps.
// LDS: bufX 32KB + bufY 32KB + kbuf 96KB = 160KB (of 320KB per WGP).
// ---------------------------------------------------------------------------
__global__ __launch_bounds__(256, 1)
void node_tsit5_kernel(const float* __restrict__ ts,
                       const float* __restrict__ y0,
                       const _Float16* __restrict__ wt,
                       const float* __restrict__ b0,
                       const float* __restrict__ b1,
                       const float* __restrict__ b2,
                       const float* __restrict__ b3,
                       float* __restrict__ out)
{
    __shared__ _Float16 bufX[64 * 256];
    __shared__ _Float16 bufY[64 * 256];
    __shared__ float    kbuf[6][64 * 64];

    const int tid  = threadIdx.x;
    const int wave = tid >> 5;
    const int lane = tid & 31;
    const int rowG0 = blockIdx.x * 64;

    const _Float16* Wt0 = wt;
    const _Float16* Wt1 = wt + OFF_WT1;
    const _Float16* Wt2 = wt + OFF_WT2;
    const _Float16* Wt3 = wt + OFF_WT3;

    // Tsit5 A-matrix (strictly lower triangular), row s = coefficients of k_j.
    const float ACO[6][5] = {
        { 0.f,  0.f,  0.f,  0.f,  0.f },
        { A21,  0.f,  0.f,  0.f,  0.f },
        { A31,  A32,  0.f,  0.f,  0.f },
        { A41,  A42,  A43,  0.f,  0.f },
        { A51,  A52,  A53,  A54,  0.f },
        { A61,  A62,  A63,  A64,  A65 },
    };

    // Element ownership (matches WMMA C/D layout of the output layer):
    // e = j*8 + r -> (row, col) inside the 64x64 slab; off64 = row*64 + col.
    const int lo = lane & 15;
    const int rowOff = ((lane >> 4) & 1) * 8;
    int off64[16];
#pragma unroll
    for (int j = 0; j < 2; ++j) {
        const int p = wave * 2 + j;
        const int m = p >> 2, n = p & 3;
#pragma unroll
        for (int r = 0; r < 8; ++r)
            off64[j * 8 + r] = (m * 16 + r + rowOff) * 64 + n * 16 + lo;
    }

    // Per-lane bias scalars, hoisted out of the time loop.
    const int n0 = wave * 32;
    const float bv00 = b0[n0 + lo],           bv01 = b0[n0 + 16 + lo];
    const float bv10 = b1[n0 + lo],           bv11 = b1[n0 + 16 + lo];
    const float bv20 = b2[n0 + lo],           bv21 = b2[n0 + 16 + lo];
    const float bv3A = b3[((wave * 2) & 3) * 16 + lo];
    const float bv3B = b3[(((wave * 2) & 3) + 1) * 16 + lo];

    float yy[16];
#pragma unroll
    for (int e = 0; e < 16; ++e) {
        const size_t gi = (size_t)rowG0 * DIM + off64[e];
        yy[e] = y0[gi];
        out[gi] = yy[e];                 // out[0] = y0
    }

    for (int t = 0; t < NT - 1; ++t) {
        const float h = ts[t + 1] - ts[t];

        for (int s = 0; s < 6; ++s) {
            // Stage input: y + h * sum_j A[s][j] * k_j  (local + LDS reads)
            float tmp[16];
#pragma unroll
            for (int e = 0; e < 16; ++e) tmp[e] = yy[e];
            for (int j = 0; j < s; ++j) {
                const float c = h * ACO[s][j];
                const float* kj = kbuf[j];
#pragma unroll
                for (int e = 0; e < 16; ++e) tmp[e] += c * kj[off64[e]];
            }
#pragma unroll
            for (int e = 0; e < 16; ++e) bufX[off64[e]] = (_Float16)tmp[e];
            __syncthreads();

            // MLP: bufX(64x64) -> bufY -> bufX -> bufY -> kbuf[s]
            layer_hidden(bufX, 64, 64,   bufY, Wt0, bv00, bv01, wave, lane);
            __syncthreads();
            layer_hidden(bufY, 256, 256, bufX, Wt1, bv10, bv11, wave, lane);
            __syncthreads();
            layer_hidden(bufX, 256, 256, bufY, Wt2, bv20, bv21, wave, lane);
            __syncthreads();
            layer_out(bufY, Wt3, bv3A, bv3B, wave, lane, kbuf[s], off64);
            __syncthreads();
        }

        // y_{t+1} = y + h * sum_i B_i k_i ; emit to global output.
#pragma unroll
        for (int e = 0; e < 16; ++e) {
            const int o = off64[e];
            const float acc = B1c * kbuf[0][o] + B2c * kbuf[1][o] +
                              B3c * kbuf[2][o] + B4c * kbuf[3][o] +
                              B5c * kbuf[4][o] + B6c * kbuf[5][o];
            yy[e] += h * acc;
            out[(size_t)(t + 1) * BATCH * DIM + (size_t)rowG0 * DIM + o] = yy[e];
        }
    }
}

// ---------------------------------------------------------------------------
extern "C" void kernel_launch(void* const* d_in, const int* in_sizes, int n_in,
                              void* d_out, int out_size, void* d_ws, size_t ws_size,
                              hipStream_t stream)
{
    const float* ts = (const float*)d_in[0];
    const float* y0 = (const float*)d_in[1];
    const float* W0 = (const float*)d_in[2];
    const float* b0 = (const float*)d_in[3];
    const float* W1 = (const float*)d_in[4];
    const float* b1 = (const float*)d_in[5];
    const float* W2 = (const float*)d_in[6];
    const float* b2 = (const float*)d_in[7];
    const float* W3 = (const float*)d_in[8];
    const float* b3 = (const float*)d_in[9];
    float* out = (float*)d_out;
    _Float16* wt = (_Float16*)d_ws;

    prep_weights_kernel<<<WT_TOTAL / 256, 256, 0, stream>>>(W0, W1, W2, W3, wt);
    node_tsit5_kernel<<<BATCH / 64, 256, 0, stream>>>(ts, y0, wt,
                                                      b0, b1, b2, b3, out);
}
